// CSC1d_46548855554752
// MI455X (gfx1250) — compile-verified, matrix operand order checked by reading, not run on hardware
//
#include <hip/hip_runtime.h>
#include <math.h>

#define BATCH 16
#define CH    8
#define KATOM 64
#define KS    64
#define TLEN  8192
#define TZ    8129
#define ZBL   8256   // 63 + TZ + 64 pad -> branchless convt indexing
#define LMBDA 0.1f

typedef __attribute__((ext_vector_type(16))) __bf16 v16bf;
typedef __attribute__((ext_vector_type(8)))  float  v8f;

union ABop { unsigned u[8]; v16bf v; };

__device__ __forceinline__ unsigned short f2bf(float f) {
  union { float f; unsigned u; } cv; cv.f = f;
  unsigned u = cv.u;
  u += 0x7fffu + ((u >> 16) & 1u);   // round-to-nearest-even
  return (unsigned short)(u >> 16);
}
__device__ __forceinline__ float bf2f(unsigned short h) {
  union { unsigned u; float f; } cv; cv.u = ((unsigned)h) << 16;
  return cv.f;
}

// ISA 7.12.2: 16-bit A (16x32) / B (32x16) reduction index for VGPR v,
// lane-half 'half', element e=0 (e=1 is the adjacent bf16 in the same dword).
__device__ __forceinline__ int kidx0(int v, int half) {
  return ((v < 4) ? 0 : 16) + half * 8 + (v & 3) * 2;
}

// ---------------------------------------------------------------------------
// One-time kernels
// ---------------------------------------------------------------------------

__global__ void zero_kernel(unsigned short* p, long n) {
  long i = (long)blockIdx.x * blockDim.x + threadIdx.x;
  long stride = (long)gridDim.x * blockDim.x;
  for (; i < n; i += stride) p[i] = 0;
}

// Dc[k][c*64+s]  = bf16(D[k,c,s])            (conv1d  A matrix, 64x512)
// Dt[m][k*64+s'] = bf16(D[k,m,63-s']) (m<8)  (convt1d A matrix, 16x4096, rows 8..15 zero)
__global__ void pack_kernel(const float* __restrict__ D,
                            unsigned short* __restrict__ Dc,
                            unsigned short* __restrict__ Dt) {
  int i = blockIdx.x * 256 + threadIdx.x;
  if (i < KATOM * CH * KS) Dc[i] = f2bf(D[i]);
  if (i < 16 * KATOM * KS) {
    int m = i >> 12, j = i & 4095, k = j >> 6, sp = j & 63;
    Dt[i] = (m < CH) ? f2bf(D[k * (CH * KS) + m * KS + (63 - sp)]) : (unsigned short)0;
  }
}

// L = sum over (k,c) rows of max_f |DFT64(D[k,c,:])|^2 ; L==0 -> 1
__global__ void lipschitz_kernel(const float* __restrict__ D, float* __restrict__ Lp) {
  __shared__ float red[256];
  const int tid = threadIdx.x;
  float local = 0.0f;
  for (int rc = tid; rc < KATOM * CH; rc += 256) {
    const float* d = D + rc * KS;
    float mx = 0.0f;
    for (int f = 0; f < KS; ++f) {
      float re = 0.0f, im = 0.0f;
      for (int s = 0; s < KS; ++s) {
        float ang = -6.28318530717958647692f * (float)(f * s) / 64.0f;
        re += d[s] * __cosf(ang);
        im += d[s] * __sinf(ang);
      }
      mx = fmaxf(mx, re * re + im * im);
    }
    local += mx;
  }
  red[tid] = local;
  __syncthreads();
  for (int off = 128; off > 0; off >>= 1) {
    if (tid < off) red[tid] += red[tid + off];
    __syncthreads();
  }
  if (tid == 0) Lp[0] = (red[0] == 0.0f) ? 1.0f : red[0];
}

// ---------------------------------------------------------------------------
// convt1d(zin, D) via WMMA: tile M=16 (c, 8 valid), wave w owns 16 t's,
// 8 waves -> 128 t's per block; reduction 4096 (k,s') in 128 chunks of 32.
// mode 0: rb = y - x (bf16);  mode 1: yout = y (f32, final output).
// ---------------------------------------------------------------------------
__global__ __launch_bounds__(256)
void convt_kernel(const unsigned short* __restrict__ zin,
                  const unsigned short* __restrict__ Dt,
                  const float* __restrict__ x,
                  unsigned short* __restrict__ rb,
                  float* __restrict__ yout,
                  int mode) {
  __shared__ unsigned short sZ[KATOM][256];   // rows padded; [0..191] used
  const int b   = blockIdx.y;
  const int t0  = blockIdx.x * 128;
  const int tid = threadIdx.x;

  // dword staging: zin[b][k][t0 .. t0+191]  (t0+191 <= 8255 = ZBL-1)
  for (int i = tid; i < KATOM * 128; i += 256) {
    int k = i >> 7, offd = i & 127;
    unsigned val = 0u;
    if (offd < 96)
      val = *(const unsigned*)(zin + (long)(b * KATOM + k) * ZBL + t0 + 2 * offd);
    ((unsigned*)&sZ[k][0])[offd] = val;
  }
  __syncthreads();

  const int lane = tid & 31;
  const int wave = tid >> 5;     // each wave: t-subtile [t0+16*wave, +15]
  const int half = lane >> 4;
  const int row  = lane & 15;

  v8f acc;
  #pragma unroll
  for (int e = 0; e < 8; ++e) acc[e] = 0.0f;

  for (int q = 0; q < 128; ++q) {          // chunk: k = q>>1, s' in (q&1)*32..+31
    const int k  = q >> 1;
    const int s0 = (q & 1) * 32;
    ABop A;
    const unsigned* dtp = (const unsigned*)(Dt + row * (KATOM * KS) + q * 32);
    #pragma unroll
    for (int v = 0; v < 8; ++v) A.u[v] = dtp[kidx0(v, half) >> 1];

    ABop Bop;
    const int toff = wave * 16 + row + s0;   // padded index: t + s' (pad 63 folded)
    #pragma unroll
    for (int v = 0; v < 8; ++v) {
      int idx = toff + kidx0(v, half);       // max 189; +1 <= 190 < 192 staged
      unsigned lo = sZ[k][idx], hi = sZ[k][idx + 1];
      Bop.u[v] = lo | (hi << 16);
    }
#if defined(__gfx1250__)
    acc = __builtin_amdgcn_wmma_f32_16x16x32_bf16(
        false, A.v, false, Bop.v, (short)0, acc, false, false);
#endif
  }

  if (half == 0) {                  // C/D rows 0..7 (valid channels) live in lanes 0-15
    const int t = t0 + wave * 16 + row;
    #pragma unroll
    for (int v = 0; v < 8; ++v) {
      const long gidx = (long)(b * CH + v) * TLEN + t;
      float y = acc[v];
      if (mode == 0) rb[gidx] = f2bf(y - x[gidx]);
      else           yout[gidx] = y;
    }
  }
}

// ---------------------------------------------------------------------------
// conv1d(r, D) + fused FISTA prox/momentum update.
// 8 waves: wave&3 -> k-tile [16*(wave&3), +15]; wave>>2 -> t half-block of 64.
// Each wave: 16 k x 64 t, reduction 512 (c,s) in 16 chunks of 32.
// ---------------------------------------------------------------------------
__global__ __launch_bounds__(256)
void conv_update_kernel(const unsigned short* __restrict__ rb,
                        const unsigned short* __restrict__ Dc,
                        const float* __restrict__ Lp,
                        unsigned short* __restrict__ zb,
                        unsigned short* __restrict__ wb,
                        float* __restrict__ zhat,
                        float gamma, int write_zhat) {
  __shared__ unsigned short sR[CH][256];   // rows padded; [0..190] used
  const int b   = blockIdx.y;
  const int t0  = blockIdx.x * 128;
  const int tid = threadIdx.x;

  // dword staging: r[b][c][t0 .. t0+191], clamped at TLEN
  for (int i = tid; i < CH * 128; i += 256) {
    int c = i >> 7, offd = i & 127;
    unsigned val = 0u;
    if (offd < 96 && (t0 + 2 * offd) < TLEN)
      val = *(const unsigned*)(rb + (long)(b * CH + c) * TLEN + t0 + 2 * offd);
    ((unsigned*)&sR[c][0])[offd] = val;
  }
  __syncthreads();

  const int lane  = tid & 31;
  const int wave  = tid >> 5;
  const int half  = lane >> 4;
  const int row   = lane & 15;
  const int kbase = (wave & 3) * 16;
  const int tblk  = (wave >> 2) * 64;

  v8f acc[4];
  #pragma unroll
  for (int n = 0; n < 4; ++n)
    #pragma unroll
    for (int e = 0; e < 8; ++e) acc[n][e] = 0.0f;

  for (int q = 0; q < 16; ++q) {           // chunk: c = q>>1, s in (q&1)*32..+31
    const int c  = q >> 1;
    const int s0 = (q & 1) * 32;
    ABop A;
    const unsigned* dcp = (const unsigned*)(Dc + (kbase + row) * (CH * KS) + q * 32);
    #pragma unroll
    for (int v = 0; v < 8; ++v) A.u[v] = dcp[kidx0(v, half) >> 1];

    #pragma unroll
    for (int nt = 0; nt < 4; ++nt) {
      ABop Bop;
      const int toff = tblk + nt * 16 + row + s0;
      #pragma unroll
      for (int v = 0; v < 8; ++v) {
        int idx = toff + kidx0(v, half);     // max 189; +1 <= 190 < 192 staged
        unsigned lo = sR[c][idx], hi = sR[c][idx + 1];
        Bop.u[v] = lo | (hi << 16);
      }
#if defined(__gfx1250__)
      acc[nt] = __builtin_amdgcn_wmma_f32_16x16x32_bf16(
          false, A.v, false, Bop.v, (short)0, acc[nt], false, false);
#endif
    }
  }

  // fused prox + momentum:  w_new = relu(z - g/L - lmbd/L);
  //                         z_new = w_new + gamma*(w_new - w)
  const float Lv   = Lp[0];
  const float Linv = 1.0f / Lv;
  const float thr  = LMBDA * Linv;
  #pragma unroll
  for (int nt = 0; nt < 4; ++nt) {
    #pragma unroll
    for (int v = 0; v < 8; ++v) {
      const int m = half * 8 + v;
      const int k = kbase + m;
      const int t = t0 + tblk + nt * 16 + row;
      if (t < TZ) {
        const long idx  = (long)(b * KATOM + k) * TZ + t;
        const long pidx = (long)(b * KATOM + k) * ZBL + 63 + t;
        float g  = acc[nt][v];
        float z  = bf2f(zb[pidx]);
        float wc = bf2f(wb[pidx]);
        float wn = fmaxf(z - g * Linv - thr, 0.0f);
        float zn = wn + gamma * (wn - wc);
        zb[pidx] = f2bf(zn);
        wb[pidx] = f2bf(wn);
        if (write_zhat) zhat[idx] = wn;
      }
    }
  }
}

// ---------------------------------------------------------------------------
extern "C" void kernel_launch(void* const* d_in, const int* in_sizes, int n_in,
                              void* d_out, int out_size, void* d_ws, size_t ws_size,
                              hipStream_t stream) {
  const float* x = (const float*)d_in[0];   // (16, 8, 8192)
  const float* D = (const float*)d_in[1];   // (64, 8, 64)

  float* out0 = (float*)d_out;                          // convt1d(z_hat, D): 16*8*8192
  float* zhat = out0 + (long)BATCH * CH * TLEN;         // z_hat: 16*64*8129

  char* ws = (char*)d_ws;
  float*          Lp = (float*)ws;                                  // 256 B
  unsigned short* Dc = (unsigned short*)(ws + 256);                 // 64x512 bf16
  unsigned short* Dt = (unsigned short*)(ws + 256 + 65536);         // 16x4096 bf16
  unsigned short* zb = (unsigned short*)(ws + 256 + 65536 + 131072);
  unsigned short* wb = zb + (long)BATCH * KATOM * ZBL;
  unsigned short* rb = wb + (long)BATCH * KATOM * ZBL;              // 16x8x8192 bf16
  // total ws use ~36.1 MB

  // z0 = w0 = 0 (zb and wb are contiguous -> one zeroing pass; halos stay 0)
  zero_kernel<<<2048, 256, 0, stream>>>(zb, 2L * BATCH * KATOM * ZBL);
  pack_kernel<<<256, 256, 0, stream>>>(D, Dc, Dt);
  lipschitz_kernel<<<1, 256, 0, stream>>>(D, Lp);

  dim3 grid(64, BATCH), blk(256);
  double beta = 1.0;
  for (int it = 0; it < 20; ++it) {
    double bn = (1.0 + sqrt(1.0 + 4.0 * beta * beta)) * 0.5;
    float gamma = (float)((beta - 1.0) / bn);   // gamma[0] = 0
    beta = bn;
    convt_kernel<<<grid, blk, 0, stream>>>(zb, Dt, x, rb, nullptr, 0);
    conv_update_kernel<<<grid, blk, 0, stream>>>(rb, Dc, Lp, zb, wb, zhat,
                                                 gamma, (it == 19) ? 1 : 0);
  }
  // final: out0 = convt1d(z_hat = w, D)
  convt_kernel<<<grid, blk, 0, stream>>>(wb, Dt, x, nullptr, out0, 1);
}
